// Attention_58110907514964
// MI455X (gfx1250) — compile-verified, matrix operand order checked by reading, not run on hardware
//
#include <hip/hip_runtime.h>

// ---------------------------------------------------------------------------
// Causal MHA forward for MI455X (gfx1250): bf16 WMMA + async global->LDS
// double-buffered staging (ASYNCcnt). f32 accumulation throughout.
// ---------------------------------------------------------------------------

typedef __attribute__((ext_vector_type(16))) __bf16          v16bf;
typedef __attribute__((ext_vector_type(16))) unsigned short  v16u;
typedef __attribute__((ext_vector_type(8)))  unsigned short  u16x8;
typedef __attribute__((ext_vector_type(8)))  float           v8f;

#define TT 2048
#define EE 1024
#define HH 16
#define HS 64

static __device__ __forceinline__ unsigned short f32_to_bf16(float f) {
    unsigned int u = __float_as_uint(f);
    u += 0x7FFFu + ((u >> 16) & 1u);   // round-to-nearest-even
    return (unsigned short)(u >> 16);
}

// Async 16-byte global -> LDS copy (per-lane), tracked by ASYNCcnt.
// VDST = LDS byte address (generic LDS pointer truncated to 32 bits),
// VADDR = 64-bit global address, SADDR = off.
static __device__ __forceinline__ void cp_b128(unsigned short* lds_dst,
                                               const unsigned short* gsrc) {
    unsigned int lds_addr = (unsigned int)(size_t)lds_dst;
    unsigned long long gaddr = (unsigned long long)(size_t)gsrc;
    asm volatile("global_load_async_to_lds_b128 %0, %1, off"
                 :
                 : "v"(lds_addr), "v"(gaddr)
                 : "memory");
}

static __device__ __forceinline__ void wait_async0() {
    asm volatile("s_wait_asynccnt 0" ::: "memory");
}

// WMMA bf16: D = A*B + C, 16x16x32, f32 accum.
static __device__ __forceinline__ v8f wmma_bf16(v16u a, v16u b, v8f c) {
    return __builtin_amdgcn_wmma_f32_16x16x32_bf16(
        false, __builtin_bit_cast(v16bf, a),
        false, __builtin_bit_cast(v16bf, b),
        (short)0, c, false, false);
}

// Load one A/B fragment for a 32-wide K-run starting at LDS row pointer p.
// ISA layout: element i holds K = (lane>>4)*8 + (i&7) + (i>=8 ? 16 : 0),
// i.e. two contiguous 8-element runs -> two ds_load_b128.
static __device__ __forceinline__ v16u load_frag32(const unsigned short* p, int kb8) {
    u16x8 lo = *(const u16x8*)(p + kb8);
    u16x8 hi = *(const u16x8*)(p + kb8 + 16);
    v16u r;
#pragma unroll
    for (int i = 0; i < 8; ++i) { r[i] = lo[i]; r[i + 8] = hi[i]; }
    return r;
}

// ---------------------------------------------------------------------------
// f32 -> bf16 conversion (grid-stride, no transpose)
// ---------------------------------------------------------------------------
__global__ __launch_bounds__(256) void cvt_bf16_kernel(const float* __restrict__ in,
                                                       unsigned short* __restrict__ out,
                                                       int n) {
    for (int i = blockIdx.x * blockDim.x + threadIdx.x; i < n; i += gridDim.x * blockDim.x)
        out[i] = f32_to_bf16(in[i]);
}

// ---------------------------------------------------------------------------
// f32 [K,N] -> bf16 transposed [N,K] (tiled through LDS, coalesced both ways)
// ---------------------------------------------------------------------------
__global__ __launch_bounds__(256) void cvt_bf16_t_kernel(const float* __restrict__ in,
                                                         unsigned short* __restrict__ out,
                                                         int K, int N) {
    __shared__ float t[32][33];
    const int tk = blockIdx.y * 32, tn = blockIdx.x * 32;
    const int c = threadIdx.x & 31, r4 = threadIdx.x >> 5;   // 8 rows per pass
#pragma unroll
    for (int rr = 0; rr < 32; rr += 8)
        t[rr + r4][c] = in[(size_t)(tk + rr + r4) * N + tn + c];
    __syncthreads();
#pragma unroll
    for (int rr = 0; rr < 32; rr += 8)
        out[(size_t)(tn + rr + r4) * K + tk + c] = f32_to_bf16(t[c][rr + r4]);
}

// ---------------------------------------------------------------------------
// Tiled bf16 GEMM:  C[M,N] = A[M,K] * Bt[N,K]^T + bias[N]
// Both operands row-copy straight into LDS (B pre-transposed on host side),
// async double-buffered K steps of 32. Block 256 = 8 waves, 128x128 tile.
// MODE 0: scatter into Q[b,h,t,64] (pre-scaled by 1/sqrt(hs)) /
//         K[b,h,t,64] / V[b,h,d,t] bf16.
// MODE 1: write f32 to `of` (out projection).
// ---------------------------------------------------------------------------
template <int MODE>
__global__ __launch_bounds__(256) void gemm_bf16_kernel(
    const unsigned short* __restrict__ A, const unsigned short* __restrict__ Bt,
    const float* __restrict__ bias,
    unsigned short* __restrict__ oq, unsigned short* __restrict__ ok,
    unsigned short* __restrict__ ov, float* __restrict__ of,
    int M, int N, int K) {
    __shared__ unsigned short As[2][128 * 40];   // [row][k], stride 40 (80B rows)
    __shared__ unsigned short Bs[2][128 * 40];   // [n][k],  stride 40

    const int tid  = threadIdx.x;
    const int lane = tid & 31;
    const int wid  = tid >> 5;
    const int ln   = lane & 15;
    const int hi   = lane >> 4;
    const int kb8  = hi * 8;
    const int rowBase = blockIdx.y * 128;
    const int colBase = blockIdx.x * 128;
    const int wm = wid >> 2;   // 0..1 : 64 rows each
    const int wn = wid & 3;    // 0..3 : 32 cols each

    auto stage = [&](int buf, int kk) {
        for (int c = tid; c < 1024; c += 256) {
            if (c < 512) {
                int row = c >> 2, seg = c & 3;
                cp_b128(&As[buf][row * 40 + seg * 8],
                        A + (size_t)(rowBase + row) * K + kk + seg * 8);
            } else {
                int c2 = c - 512, n = c2 >> 2, seg = c2 & 3;
                cp_b128(&Bs[buf][n * 40 + seg * 8],
                        Bt + (size_t)(colBase + n) * K + kk + seg * 8);
            }
        }
    };

    v8f acc[4][2];
#pragma unroll
    for (int mt = 0; mt < 4; ++mt)
#pragma unroll
        for (int nt = 0; nt < 2; ++nt) acc[mt][nt] = v8f{};

    stage(0, 0);
    wait_async0();
    __syncthreads();

    int buf = 0;
    for (int kk = 0; kk < K; kk += 32) {
        if (kk + 32 < K) stage(buf ^ 1, kk + 32);   // overlap with compute below

        v16u af[4], bf[2];
#pragma unroll
        for (int mt = 0; mt < 4; ++mt)
            af[mt] = load_frag32(&As[buf][(wm * 64 + mt * 16 + ln) * 40], kb8);
#pragma unroll
        for (int nt = 0; nt < 2; ++nt)
            bf[nt] = load_frag32(&Bs[buf][(wn * 32 + nt * 16 + ln) * 40], kb8);
#pragma unroll
        for (int mt = 0; mt < 4; ++mt)
#pragma unroll
            for (int nt = 0; nt < 2; ++nt)
                acc[mt][nt] = wmma_bf16(af[mt], bf[nt], acc[mt][nt]);

        wait_async0();
        __syncthreads();
        buf ^= 1;
    }

    // Epilogue: C element r -> row = r + 8*hi, col = lane&15.
#pragma unroll
    for (int mt = 0; mt < 4; ++mt)
#pragma unroll
        for (int nt = 0; nt < 2; ++nt)
#pragma unroll
            for (int r = 0; r < 8; ++r) {
                int row = rowBase + wm * 64 + mt * 16 + r + hi * 8;
                int col = colBase + wn * 32 + nt * 16 + ln;
                float val = acc[mt][nt][r] + bias[col];
                if (MODE == 0) {
                    int which = col >> 10, e = col & 1023;
                    int h = e >> 6, d = e & 63;
                    int b = row >> 11, t = row & 2047;
                    if (which == 0)   // fold softmax scale 1/sqrt(64) into Q
                        oq[(((size_t)(b * HH + h)) * TT + t) * HS + d] =
                            f32_to_bf16(val * 0.125f);
                    else if (which == 1)
                        ok[(((size_t)(b * HH + h)) * TT + t) * HS + d] = f32_to_bf16(val);
                    else  // V stored transposed: [b,h,d,t]
                        ov[(((size_t)(b * HH + h)) * HS + d) * TT + t] = f32_to_bf16(val);
                } else {
                    of[(size_t)row * N + col] = val;
                }
            }
}

// ---------------------------------------------------------------------------
// Flash attention: grid (T/64, B*H), 128 threads = 4 waves.
// Each wave owns 16 query rows; causal K/V tiles of 64, async double-buffered.
// K is [b,h,t,d]; V is pre-transposed [b,h,d,t] so both stages are row copies.
// Q arrives pre-scaled by 1/sqrt(hs); only the diagonal tile needs masking.
// ---------------------------------------------------------------------------
__global__ __launch_bounds__(128) void attn_kernel(const unsigned short* __restrict__ q,
                                                   const unsigned short* __restrict__ k,
                                                   const unsigned short* __restrict__ v,
                                                   unsigned short* __restrict__ o) {
    __shared__ unsigned short Qs[64 * 72];        // [qrow][d]
    __shared__ unsigned short Ks[2][64 * 72];     // [key][d]   (B-frag for Q*K^T)
    __shared__ unsigned short Vs[2][64 * 72];     // [d][key]   (B-frag for P*V)
    __shared__ unsigned short Ps[4 * 16 * 72];    // per-wave P tile [row][key]

    const int tid  = threadIdx.x;
    const int lane = tid & 31;
    const int wid  = tid >> 5;
    const int ln   = lane & 15;
    const int hif  = lane >> 4;
    const int kb8  = hif * 8;
    const int qt   = blockIdx.x;                  // query tile (64 rows)
    const int bh   = blockIdx.y;                  // b*16 + h
    const size_t bhT = (size_t)bh * TT * HS;      // base for q/k ([t][d])
    const size_t bhD = (size_t)bh * HS * TT;      // base for v  ([d][t])

    auto stage_kv = [&](int buf, int kb0) {
        for (int c = tid; c < 1024; c += 128) {
            if (c < 512) {
                int row = c >> 3, seg = c & 7;
                cp_b128(&Ks[buf][row * 72 + seg * 8],
                        k + bhT + (size_t)(kb0 + row) * HS + seg * 8);
            } else {
                int c2 = c - 512, d = c2 >> 3, seg = c2 & 7;
                cp_b128(&Vs[buf][d * 72 + seg * 8],
                        v + bhD + (size_t)d * TT + kb0 + seg * 8);
            }
        }
    };

    // Stage Q tile + first K/V tile.
    for (int c = tid; c < 512; c += 128) {
        int row = c >> 3, seg = c & 7;
        cp_b128(&Qs[row * 72 + seg * 8],
                q + bhT + (size_t)(qt * 64 + row) * HS + seg * 8);
    }
    stage_kv(0, 0);
    wait_async0();
    __syncthreads();

    v16u qf[2];
#pragma unroll
    for (int kc = 0; kc < 2; ++kc)
        qf[kc] = load_frag32(Qs + (wid * 16 + ln) * 72 + kc * 32, kb8);

    float mrow[8], lrow[8];
    v8f   oacc[4];
#pragma unroll
    for (int r = 0; r < 8; ++r) { mrow[r] = -3.0e38f; lrow[r] = 0.0f; }
#pragma unroll
    for (int n = 0; n < 4; ++n) oacc[n] = v8f{};

    int buf = 0;
    for (int kt = 0; kt <= qt; ++kt) {
        const int kb0 = kt * 64;
        if (kt < qt) stage_kv(buf ^ 1, kb0 + 64);   // overlap next tile

        // S = Q * K^T (16 x 64 per wave); kc-outer so the 4 WMMAs are independent.
        v8f s[4];
#pragma unroll
        for (int nt = 0; nt < 4; ++nt) s[nt] = v8f{};
#pragma unroll
        for (int kc = 0; kc < 2; ++kc)
#pragma unroll
            for (int nt = 0; nt < 4; ++nt) {
                v16u kfrag = load_frag32(&Ks[buf][(nt * 16 + ln) * 72 + kc * 32], kb8);
                s[nt] = wmma_bf16(qf[kc], kfrag, s[nt]);
            }

        // Causal mask: only the diagonal tile has any masked entries.
        if (kt == qt) {
            const int qg0 = qt * 64 + wid * 16 + hif * 8;
#pragma unroll
            for (int nt = 0; nt < 4; ++nt) {
                int keyg = kb0 + nt * 16 + ln;
#pragma unroll
                for (int r = 0; r < 8; ++r)
                    if (keyg > qg0 + r) s[nt][r] = -1.0e30f;
            }
        }

        // Online softmax; a row's 16 columns live in one 16-lane half,
        // so xor-shuffles with width 16 reduce exactly one row.
#pragma unroll
        for (int r = 0; r < 8; ++r) {
            float vmax = fmaxf(fmaxf(s[0][r], s[1][r]), fmaxf(s[2][r], s[3][r]));
            vmax = fmaxf(vmax, __shfl_xor(vmax, 1, 16));
            vmax = fmaxf(vmax, __shfl_xor(vmax, 2, 16));
            vmax = fmaxf(vmax, __shfl_xor(vmax, 4, 16));
            vmax = fmaxf(vmax, __shfl_xor(vmax, 8, 16));
            float mnew = fmaxf(mrow[r], vmax);
            float corr = __expf(mrow[r] - mnew);
            float rsum = 0.0f;
#pragma unroll
            for (int nt = 0; nt < 4; ++nt) {
                float p = __expf(s[nt][r] - mnew);
                s[nt][r] = p;
                rsum += p;
            }
            rsum += __shfl_xor(rsum, 1, 16);
            rsum += __shfl_xor(rsum, 2, 16);
            rsum += __shfl_xor(rsum, 4, 16);
            rsum += __shfl_xor(rsum, 8, 16);
            lrow[r] = lrow[r] * corr + rsum;
            mrow[r] = mnew;
#pragma unroll
            for (int n = 0; n < 4; ++n) oacc[n][r] *= corr;
        }

        // Re-layout P (C-frag) -> A-frag via wave-private LDS.
        unsigned short* pw = Ps + wid * 16 * 72;
#pragma unroll
        for (int nt = 0; nt < 4; ++nt)
#pragma unroll
            for (int r = 0; r < 8; ++r)
                pw[(r + hif * 8) * 72 + nt * 16 + ln] = f32_to_bf16(s[nt][r]);
        asm volatile("s_wait_dscnt 0" ::: "memory");   // wave-local store->load fence

#pragma unroll
        for (int kc = 0; kc < 2; ++kc) {
            v16u pf = load_frag32(pw + ln * 72 + kc * 32, kb8);
#pragma unroll
            for (int n = 0; n < 4; ++n) {
                v16u vfrag = load_frag32(&Vs[buf][(n * 16 + ln) * 72 + kc * 32], kb8);
                oacc[n] = wmma_bf16(pf, vfrag, oacc[n]);
            }
        }

        wait_async0();
        __syncthreads();
        buf ^= 1;
    }

    // Write O / l as bf16 into [b, t, h*64 + d] layout.
    const int b = bh >> 4, h = bh & 15;
#pragma unroll
    for (int n = 0; n < 4; ++n)
#pragma unroll
        for (int r = 0; r < 8; ++r) {
            int qrow = qt * 64 + wid * 16 + r + hif * 8;
            int d    = n * 16 + ln;
            float val = oacc[n][r] / lrow[r];
            o[(((size_t)b * TT + qrow) * EE) + h * HS + d] = f32_to_bf16(val);
        }
}

// ---------------------------------------------------------------------------
// Host launch
// ---------------------------------------------------------------------------
extern "C" void kernel_launch(void* const* d_in, const int* in_sizes, int n_in,
                              void* d_out, int out_size, void* d_ws, size_t ws_size,
                              hipStream_t stream) {
    (void)in_sizes; (void)n_in; (void)out_size; (void)ws_size;
    const float* x    = (const float*)d_in[0];
    const float* Wqkv = (const float*)d_in[1];
    const float* bqkv = (const float*)d_in[2];
    const float* Wout = (const float*)d_in[3];
    const float* bout = (const float*)d_in[4];
    float* out = (float*)d_out;

    unsigned short* ws     = (unsigned short*)d_ws;
    unsigned short* xb     = ws;                                   // [8192,1024]
    unsigned short* WqkvbT = xb     + (size_t)8192 * 1024;         // [3072,1024]
    unsigned short* WoutbT = WqkvbT + (size_t)3072 * 1024;         // [1024,1024]
    unsigned short* qb     = WoutbT + (size_t)1024 * 1024;         // [b,h,t,64]
    unsigned short* kb     = qb     + (size_t)8388608;             // [b,h,t,64]
    unsigned short* vb     = kb     + (size_t)8388608;             // [b,h,64,t]
    unsigned short* ab     = vb     + (size_t)8388608;             // [8192,1024]

    cvt_bf16_kernel<<<2048, 256, 0, stream>>>(x, xb, 8192 * 1024);
    cvt_bf16_t_kernel<<<dim3(96, 32), 256, 0, stream>>>(Wqkv, WqkvbT, 1024, 3072);
    cvt_bf16_t_kernel<<<dim3(32, 32), 256, 0, stream>>>(Wout, WoutbT, 1024, 1024);

    // QKV projection: [8192,1024] x [1024,3072] -> scatter Q/K/V bf16.
    gemm_bf16_kernel<0><<<dim3(24, 64), 256, 0, stream>>>(
        xb, WqkvbT, bqkv, qb, kb, vb, nullptr, 8192, 3072, 1024);

    // Causal flash attention per (batch*head, 64-query tile).
    attn_kernel<<<dim3(32, 64), 128, 0, stream>>>(qb, kb, vb, ab);

    // Output projection: [8192,1024] x [1024,1024] -> f32 out + bias.
    gemm_bf16_kernel<1><<<dim3(8, 64), 256, 0, stream>>>(
        ab, WoutbT, bout, nullptr, nullptr, nullptr, out, 8192, 1024, 1024);
}
